// GraphAttention_50259707298082
// MI455X (gfx1250) — matrix operand order repaired and checked by reading, not compiled, not run on hardware
//
#include <hip/hip_runtime.h>
#include <stdint.h>

typedef __attribute__((ext_vector_type(16))) __bf16 v16bf;
typedef __attribute__((ext_vector_type(8)))  float  v8f;

#define Bsz   32
#define Nsz   1024
#define Csz   128
#define NEGV  (-9000000000000000.0f)
#define ALPHA 0.2f

__device__ __forceinline__ unsigned short f2bf(float f) {
    unsigned u = __float_as_uint(f);
    u += 0x7FFFu + ((u >> 16) & 1u);          // round-to-nearest-even
    return (unsigned short)(u >> 16);
}

// ---------------- Kernel 1: w1 = W @ a[:E], w2 = W @ a[E:]  (128x128 * 128) ----
__global__ void prep_w(const float* __restrict__ W, const float* __restrict__ a,
                       float* __restrict__ w1, float* __restrict__ w2) {
    int c = threadIdx.x;                       // 128 threads
    const float* row = W + (size_t)c * Csz;
    float a1 = 0.f, a2 = 0.f;
    for (int e = 0; e < Csz; ++e) { float wv = row[e]; a1 += wv * a[e]; a2 += wv * a[Csz + e]; }
    w1[c] = a1; w2[c] = a2;
}

// ---------------- Kernel 2: s1/s2 = x @ w1 / x @ w2 (one wave per row) --------
__global__ void node_scores(const float* __restrict__ x, const float* __restrict__ w1,
                            const float* __restrict__ w2, float* __restrict__ s1,
                            float* __restrict__ s2) {
    int wave = threadIdx.x >> 5, lane = threadIdx.x & 31;
    int row  = blockIdx.x * 8 + wave;          // 4096 blocks * 8 waves = 32768 rows
    float4 xv  = ((const float4*)(x + (size_t)row * Csz))[lane];   // 512B coalesced per wave
    float4 w1v = ((const float4*)w1)[lane];
    float4 w2v = ((const float4*)w2)[lane];
    float d1 = xv.x * w1v.x + xv.y * w1v.y + xv.z * w1v.z + xv.w * w1v.w;
    float d2 = xv.x * w2v.x + xv.y * w2v.y + xv.z * w2v.z + xv.w * w2v.w;
    for (int off = 16; off; off >>= 1) {
        d1 += __shfl_xor(d1, off, 32);
        d2 += __shfl_xor(d2, off, 32);
    }
    if (lane == 0) { s1[row] = d1; s2[row] = d2; }
}

// ---------------- Kernel 3: xT[b][c][j] = bf16(x[b][j][c])  (LDS tile transpose)
__global__ void xpose_bf16(const float* __restrict__ x, unsigned short* __restrict__ xT) {
    __shared__ unsigned short T[Csz][66];      // stride 66 -> 33-dword pitch, conflict-free
    int b = blockIdx.y, j0 = blockIdx.x * 64;
    for (int idx = threadIdx.x; idx < 64 * Csz; idx += 256) {
        int j = idx >> 7, c = idx & 127;       // consecutive tid -> consecutive c (coalesced)
        T[c][j] = f2bf(x[((size_t)b * Nsz + j0 + j) * Csz + c]);
    }
    __syncthreads();
    for (int idx = threadIdx.x; idx < Csz * 64; idx += 256) {
        int c = idx >> 6, j = idx & 63;        // consecutive tid -> consecutive j (coalesced)
        xT[((size_t)b * Csz + c) * Nsz + j0 + j] = T[c][j];
    }
}

// ---------------- Kernel 4: masked softmax + bf16 WMMA GEMM + tanh ------------
// LDS layout (dynamic):
//   S      : 16 x 1024 f32 scores            @ 0       (65536 B)
//   P      : 16 x 1032 bf16 exp values       @ 65536   (33024 B, padded stride)
//   s1row  : 1024 f32                        @ 98560   (4096 B)
//   s2row  : 1024 f32                        @ 102656  (4096 B)
//   inv    : 16 f32 row 1/sum                @ 106752  (64 B)
#define SMEM_BYTES 106816

__global__ void __launch_bounds__(256, 1)
gat_main(const int* __restrict__ adj, const float* __restrict__ s1,
         const float* __restrict__ s2, const unsigned short* __restrict__ xT,
         float* __restrict__ out) {
    extern __shared__ __align__(16) char smem[];
    float*          S     = (float*)smem;
    unsigned short* P     = (unsigned short*)(smem + 65536);
    float*          s1row = (float*)(smem + 98560);
    float*          s2row = (float*)(smem + 102656);
    float*          inv   = (float*)(smem + 106752);

    const int b    = blockIdx.y;
    const int i0   = blockIdx.x * 16;
    const int tid  = threadIdx.x;
    const int wave = tid >> 5, lane = tid & 31;

    for (int j = tid; j < Nsz; j += 256) {
        s1row[j] = s1[b * Nsz + j];
        s2row[j] = s2[b * Nsz + j];
    }
    __syncthreads();

    // ---- score + softmax pass: wave w owns rows 2w, 2w+1 ----
    for (int rr = 0; rr < 2; ++rr) {
        const int ii = wave * 2 + rr;
        const int i  = i0 + ii;
        const int* arow = adj + ((size_t)b * Nsz + i) * Nsz;
        const bool low = (i < 512);
        float sLo = 0.f, sHi = 0.f;
        if (low) {              // idx1 == idx2 == 2i + (j>=512)
            sLo = s1row[2 * i]     + s2row[2 * i];
            sHi = s1row[2 * i + 1] + s2row[2 * i + 1];
        }
        float m = -3.0e38f;
        for (int jc = 0; jc < 32; ++jc) {
            int j = jc * 32 + lane;
            float v;
            if (low) v = (j >= 512) ? sHi : sLo;
            else { int jj = (2 * j) & (Nsz - 1); v = s1row[jj] + s2row[jj + 1]; }
            v = (v > 0.f) ? v : ALPHA * v;                 // LeakyReLU
            v = (arow[j] > 0) ? v : NEGV;                  // adjacency mask
            S[ii * Nsz + j] = v;
            m = fmaxf(m, v);
        }
        for (int off = 16; off; off >>= 1) m = fmaxf(m, __shfl_xor(m, off, 32));
        float sum = 0.f;
        for (int jc = 0; jc < 32; ++jc) {
            int j = jc * 32 + lane;
            float e = __expf(S[ii * Nsz + j] - m);
            sum += e;
            P[ii * 1032 + j] = f2bf(e);
        }
        for (int off = 16; off; off >>= 1) sum += __shfl_xor(sum, off, 32);
        if (lane == 0) inv[ii] = 1.f / sum;
    }
    __syncthreads();

    // ---- GEMM: out_tile(16x16) per wave, K = 1024 in steps of 32 (bf16 WMMA) ----
    const int c0 = wave * 16;
    const int n  = lane & 15;
    const int hi = lane >> 4;
    v8f acc = {0.f, 0.f, 0.f, 0.f, 0.f, 0.f, 0.f, 0.f};
    // A-frag (ISA 16-bit A 16x32): lanes0-15: K j0+0..7 / j0+16..23; lanes16-31: +8 / +24
    const char* Pbase = (const char*)P + (size_t)n * 2064 + (hi ? 16 : 0);
    // B-frag (ISA 16-bit B 32x16): lanes0-15: K 0..7 / 16..23; lanes16-31: +8 / +24, col = n
    const unsigned short* xrow = xT + ((size_t)b * Csz + c0 + n) * Nsz + (hi ? 8 : 0);

    union Frag { uint4 q[2]; v16bf v; };
    for (int j0 = 0; j0 < Nsz; j0 += 32) {
        Frag af, bf;
        af.q[0] = *(const uint4*)(Pbase + 2 * j0);        // 16B LDS, 16B-aligned
        af.q[1] = *(const uint4*)(Pbase + 2 * j0 + 32);
        bf.q[0] = *(const uint4*)(xrow + j0);             // 32B contiguous global (L2)
        bf.q[1] = *(const uint4*)(xrow + j0 + 16);
        acc = __builtin_amdgcn_wmma_f32_16x16x32_bf16(false, af.v, false, bf.v,
                                                      (short)0, acc, false, false);
    }

    // C/D layout: VGPR r -> M = r + 8*hi, N = n
    for (int r = 0; r < 8; ++r) {
        int M = r + hi * 8;
        float val = tanhf(acc[r] * inv[M]);
        out[((size_t)b * Nsz + i0 + M) * Csz + c0 + n] = val;
    }
}

extern "C" void kernel_launch(void* const* d_in, const int* in_sizes, int n_in,
                              void* d_out, int out_size, void* d_ws, size_t ws_size,
                              hipStream_t stream) {
    const float* x   = (const float*)d_in[0];   // (32,1024,128) f32
    const int*   adj = (const int*)d_in[1];     // (32,1024,1024) i32
    const float* W   = (const float*)d_in[2];   // (128,128) f32
    const float* a   = (const float*)d_in[3];   // (256,1) f32
    float*       out = (float*)d_out;           // (32,1024,128) f32

    // workspace: w1/w2 (1KB) | s1 (128KB) | s2 (128KB) | xT bf16 (8MB)  ~8.5MB total
    char* ws = (char*)d_ws;
    float*          w1 = (float*)ws;
    float*          w2 = (float*)(ws + 512);
    float*          s1 = (float*)(ws + 4096);
    float*          s2 = (float*)(ws + 4096 + 131072);
    unsigned short* xT = (unsigned short*)(ws + 524288);

    prep_w<<<1, 128, 0, stream>>>(W, a, w1, w2);
    node_scores<<<4096, 256, 0, stream>>>(x, w1, w2, s1, s2);
    dim3 gx(Nsz / 64, Bsz);
    xpose_bf16<<<gx, 256, 0, stream>>>(x, xT);
    dim3 gm(Nsz / 16, Bsz);
    gat_main<<<gm, 256, SMEM_BYTES, stream>>>(adj, s1, s2, xT, out);
}
